// OTStyleModel_31568009626221
// MI455X (gfx1250) — compile-verified, matrix operand order checked by reading, not run on hardware
//
#include <hip/hip_runtime.h>
#include <hip/hip_bf16.h>

typedef __attribute__((ext_vector_type(16))) __bf16 v16bf;
typedef __attribute__((ext_vector_type(8)))  __bf16 v8bf;
typedef __attribute__((ext_vector_type(8)))  float  v8f;

#define BIG 1e9f

// ---------------------------------------------------------------------------
// Split an fp32 into bf16 hi (exact truncation) + bf16 lo (residual, RTE).
// hi*hi + hi*lo + lo*hi reproduces the fp32 product to ~2^-16 relative.
// ---------------------------------------------------------------------------
__device__ __forceinline__ void bf16_split(float f, __bf16& hi, __bf16& lo) {
    float fh = __uint_as_float(__float_as_uint(f) & 0xFFFF0000u);
    hi = (__bf16)fh;          // exact: low mantissa bits already zero
    lo = (__bf16)(f - fh);    // residual, rounded to nearest even
}

// ---------------------------------------------------------------------------
// Pack V = concat(v8, v32, v128) zero-padded to 176 rows -> bf16 hi/lo planes
// ---------------------------------------------------------------------------
__global__ void pack_v_kernel(const float* __restrict__ v8,
                              const float* __restrict__ v32,
                              const float* __restrict__ v128,
                              __bf16* __restrict__ vhi,
                              __bf16* __restrict__ vlo) {
    int idx = blockIdx.x * 256 + threadIdx.x;
    if (idx >= 176 * 768) return;
    int row = idx / 768, col = idx - row * 768;
    float f = 0.f;
    if (row < 8)        f = v8 [row * 768 + col];
    else if (row < 40)  f = v32[(row - 8) * 768 + col];
    else if (row < 168) f = v128[(row - 40) * 768 + col];
    __bf16 h, l; bf16_split(f, h, l);
    vhi[idx] = h; vlo[idx] = l;
}

// ---------------------------------------------------------------------------
// Pack W2 = concat(Wc[192,1536], Ws[128,1536]) -> bf16 hi/lo planes
// ---------------------------------------------------------------------------
__global__ void pack_w_kernel(const float* __restrict__ Wc,
                              const float* __restrict__ Ws,
                              __bf16* __restrict__ whi,
                              __bf16* __restrict__ wlo) {
    int idx = blockIdx.x * 256 + threadIdx.x;
    if (idx >= 320 * 1536) return;
    int row = idx / 1536, col = idx - row * 1536;
    float f = (row < 192) ? Wc[row * 1536 + col] : Ws[(row - 192) * 1536 + col];
    __bf16 h, l; bf16_split(f, h, l);
    whi[idx] = h; wlo[idx] = l;
}

// ---------------------------------------------------------------------------
// Per-batch masked means, rep = [H[b,0,:], mean0-mean1], mcnt = min(n0,n1)
// ---------------------------------------------------------------------------
__global__ void prep_kernel(const float* __restrict__ H,
                            const int* __restrict__ tt,
                            const int* __restrict__ am,
                            float* __restrict__ rep,
                            int* __restrict__ mcnt) {
    const int b = blockIdx.x, tid = threadIdx.x;
    __shared__ float f0s[128], f1s[128];
    __shared__ int c0[128], c1[128];
    if (tid < 128) {
        int t = tt[b * 128 + tid], a = am[b * 128 + tid];
        int m0 = (t == 0) && (a == 1);
        int m1 = (t == 1) && (a == 1);
        f0s[tid] = (float)m0; f1s[tid] = (float)m1;
        c0[tid] = m0; c1[tid] = m1;
    }
    __syncthreads();
    for (int s = 64; s > 0; s >>= 1) {
        if (tid < s) { c0[tid] += c0[tid + s]; c1[tid] += c1[tid + s]; }
        __syncthreads();
    }
    const float inv0 = 1.f / fmaxf((float)c0[0], 1.f);
    const float inv1 = 1.f / fmaxf((float)c1[0], 1.f);
    if (tid == 0) mcnt[b] = (c0[0] < c1[0]) ? c0[0] : c1[0];
    const float* Hb = H + (size_t)b * 128 * 768;
    for (int dd = tid; dd < 768; dd += 256) {
        float s0 = 0.f, s1 = 0.f;
        for (int l = 0; l < 128; ++l) {
            float h = Hb[(size_t)l * 768 + dd];
            s0 += h * f0s[l]; s1 += h * f1s[l];
        }
        rep[(size_t)b * 1536 + dd]       = Hb[dd];                // H[b,0,:]
        rep[(size_t)b * 1536 + 768 + dd] = s0 * inv0 - s1 * inv1; // mean0-mean1
    }
}

// ---------------------------------------------------------------------------
// WMMA GEMM:  C = A[M,K](f32) @ B[N,K](bf16 hi/lo, row-major)^T
// bf16-split: 3x v_wmma_f32_16x16x32_bf16 per K-step of 32.
// LDA/LDB are COMPILE-TIME so every per-tile B offset (nt*16*LDB) folds into
// the global_load_b128 24-bit immediate -> one loop-carried base pointer per
// operand, no per-iteration address VALU, no spilled pointer table.
// k outer, n inner: A (H) loaded+split once per k-step (single HBM stream).
// blockIdx.y shifts the column window by NT*16 (keeps NT small -> no spills).
// TC=true stores C transposed for coalesced column reads downstream.
// ---------------------------------------------------------------------------
template <int NT, int LDA, int LDB, bool TC>
__global__ void wmma_gemm_split_kernel(const float* __restrict__ A,
                                       const __bf16* __restrict__ Bhi,
                                       const __bf16* __restrict__ Blo,
                                       float* __restrict__ C,
                                       int Mtiles, int Ksteps, size_t ldc) {
    const int wave = threadIdx.x >> 5;
    const int lane = threadIdx.x & 31;
    const int mt = blockIdx.x * (blockDim.x >> 5) + wave;
    if (mt >= Mtiles) return;                       // uniform per wave
    const int r16   = lane & 15;
    const int khalf = (lane >> 4) << 3;             // 0 or 8
    const int nbase = blockIdx.y * (NT * 16);
    const float*  __restrict__ aptr = A   + (size_t)(mt * 16 + r16) * LDA + khalf;
    const __bf16* __restrict__ bhp  = Bhi + (size_t)(nbase + r16) * LDB + khalf;
    const __bf16* __restrict__ blp  = Blo + (size_t)(nbase + r16) * LDB + khalf;
    const int rowbase = mt * 16 + ((lane >> 4) << 3);

    v8f acc[NT];
    #pragma unroll
    for (int nt = 0; nt < NT; ++nt) acc[nt] = {};

    for (int kt = 0; kt < Ksteps; ++kt) {
        const int k0 = kt * 32;
        __builtin_prefetch(aptr + k0 + 32, 0, 0);   // global_prefetch_b8 (next k-step)
        // ---- load + split A tile once per k-step ----
        alignas(16) float av[16];
        *(float4*)(av + 0)  = *(const float4*)(aptr + k0 + 0);
        *(float4*)(av + 4)  = *(const float4*)(aptr + k0 + 4);
        *(float4*)(av + 8)  = *(const float4*)(aptr + k0 + 16);
        *(float4*)(av + 12) = *(const float4*)(aptr + k0 + 20);
        v16bf ahi, alo;
        #pragma unroll
        for (int i = 0; i < 16; ++i) {
            __bf16 h, l; bf16_split(av[i], h, l);
            ahi[i] = h; alo[i] = l;
        }
        // ---- sweep NT column tiles against this A tile ----
        #pragma unroll
        for (int nt = 0; nt < NT; ++nt) {
            // nt*16*LDB is a compile-time constant -> immediate offset in the load
            v8bf bh0 = *(const v8bf*)(bhp + nt * 16 * LDB + k0);
            v8bf bh1 = *(const v8bf*)(bhp + nt * 16 * LDB + k0 + 16);
            v8bf bl0 = *(const v8bf*)(blp + nt * 16 * LDB + k0);
            v8bf bl1 = *(const v8bf*)(blp + nt * 16 * LDB + k0 + 16);
            v16bf bhiV, bloV;
            #pragma unroll
            for (int i = 0; i < 8; ++i) {
                bhiV[i] = bh0[i]; bhiV[8 + i] = bh1[i];
                bloV[i] = bl0[i]; bloV[8 + i] = bl1[i];
            }
            acc[nt] = __builtin_amdgcn_wmma_f32_16x16x32_bf16(false, ahi, false, bhiV,
                                                              (short)0, acc[nt], false, false);
            acc[nt] = __builtin_amdgcn_wmma_f32_16x16x32_bf16(false, ahi, false, bloV,
                                                              (short)0, acc[nt], false, false);
            acc[nt] = __builtin_amdgcn_wmma_f32_16x16x32_bf16(false, alo, false, bhiV,
                                                              (short)0, acc[nt], false, false);
        }
    }
    // ---- epilogue ----
    #pragma unroll
    for (int nt = 0; nt < NT; ++nt) {
        const int col = nbase + nt * 16 + r16;
        if (TC) {
            // transposed store: lane owns 8 consecutive rows of one column
            float* cp = C + (size_t)col * ldc + rowbase;
            *(float4*)(cp + 0) = *((const float4*)&acc[nt] + 0);
            *(float4*)(cp + 4) = *((const float4*)&acc[nt] + 1);
        } else {
            #pragma unroll
            for (int r = 0; r < 8; ++r)
                C[(size_t)(rowbase + r) * ldc + col] = acc[nt][r];
        }
    }
}

// ---------------------------------------------------------------------------
// Per-(batch, projection) bitonic sort of masked columns + |s0-s1| partial sum
// PT is the transposed projection matrix [176][32768] -> coalesced loads.
// ---------------------------------------------------------------------------
__global__ void sort_w1_kernel(const float* __restrict__ PT,
                               const int* __restrict__ tt,
                               const int* __restrict__ am,
                               const int* __restrict__ mcnt,
                               float* __restrict__ dsum) {
    const int p = blockIdx.x;    // 0..167
    const int b = blockIdx.y;    // 0..255
    const int tid = threadIdx.x; // 0..127
    __shared__ float s0[128], s1[128], red[128];
    int t = tt[b * 128 + tid], a = am[b * 128 + tid];
    bool m0 = (t == 0) && (a == 1);
    bool m1 = (t == 1) && (a == 1);
    float val = PT[(size_t)p * 32768 + b * 128 + tid];
    s0[tid] = m0 ? val : BIG;
    s1[tid] = m1 ? val : BIG;
    __syncthreads();
    for (int k = 2; k <= 128; k <<= 1) {
        for (int j = k >> 1; j > 0; j >>= 1) {
            int ixj = tid ^ j;
            if (ixj > tid) {
                bool up = ((tid & k) == 0);
                float x = s0[tid], y = s0[ixj];
                if (up ? (x > y) : (x < y)) { s0[tid] = y; s0[ixj] = x; }
                x = s1[tid]; y = s1[ixj];
                if (up ? (x > y) : (x < y)) { s1[tid] = y; s1[ixj] = x; }
            }
            __syncthreads();
        }
    }
    float d = (tid < mcnt[b]) ? fabsf(s0[tid] - s1[tid]) : 0.f;
    red[tid] = d; __syncthreads();
    for (int s = 64; s > 0; s >>= 1) {
        if (tid < s) red[tid] += red[tid + s];
        __syncthreads();
    }
    if (tid == 0) dsum[b * 168 + p] = red[0];   // deterministic (no atomics)
}

// ---------------------------------------------------------------------------
// Fuse: gate, d_ot normalization, layernorm(323), 2-way classifier
// ---------------------------------------------------------------------------
__global__ void finalize_kernel(const float* __restrict__ CS,
                                const float* __restrict__ bc,
                                const float* __restrict__ bs,
                                const float* __restrict__ gate,
                                const float* __restrict__ ln_g,
                                const float* __restrict__ ln_b,
                                const float* __restrict__ Wcls,
                                const float* __restrict__ bcls,
                                const float* __restrict__ dsum,
                                const int* __restrict__ mcnt,
                                float* __restrict__ out) {
    const int b = blockIdx.x, tid = threadIdx.x;  // blockDim = 512
    __shared__ float feat[323];
    __shared__ float red[512];
    const float g = 1.f / (1.f + expf(-gate[0]));
    if (tid < 192) {
        feat[tid] = (CS[(size_t)b * 320 + tid] + bc[tid]) * (1.f - g);
    } else if (tid < 320) {
        feat[tid] = (CS[(size_t)b * 320 + tid] + bs[tid - 192]) * g;
    } else if (tid < 323) {
        int j = tid - 320;
        int base = (j == 0) ? 0 : (j == 1) ? 8 : 40;
        int cnt  = (j == 0) ? 8 : (j == 1) ? 32 : 128;
        float s = 0.f;
        for (int i = 0; i < cnt; ++i) s += dsum[b * 168 + base + i];
        feat[tid] = s / (fmaxf((float)mcnt[b], 1.f) * (float)cnt);
    }
    __syncthreads();
    red[tid] = (tid < 323) ? feat[tid] : 0.f; __syncthreads();
    for (int s = 256; s > 0; s >>= 1) { if (tid < s) red[tid] += red[tid + s]; __syncthreads(); }
    float mu = red[0] * (1.f / 323.f);
    __syncthreads();
    float dv = (tid < 323) ? (feat[tid] - mu) : 0.f;
    red[tid] = dv * dv; __syncthreads();
    for (int s = 256; s > 0; s >>= 1) { if (tid < s) red[tid] += red[tid + s]; __syncthreads(); }
    float rstd = rsqrtf(red[0] * (1.f / 323.f) + 1e-5f);
    __syncthreads();
    if (tid < 323) feat[tid] = (feat[tid] - mu) * rstd * ln_g[tid] + ln_b[tid];
    __syncthreads();
    red[tid] = (tid < 323) ? feat[tid] * Wcls[tid] : 0.f; __syncthreads();
    for (int s = 256; s > 0; s >>= 1) { if (tid < s) red[tid] += red[tid + s]; __syncthreads(); }
    if (tid == 0) out[b * 2 + 0] = red[0] + bcls[0];
    __syncthreads();
    red[tid] = (tid < 323) ? feat[tid] * Wcls[323 + tid] : 0.f; __syncthreads();
    for (int s = 256; s > 0; s >>= 1) { if (tid < s) red[tid] += red[tid + s]; __syncthreads(); }
    if (tid == 0) out[b * 2 + 1] = red[0] + bcls[1];
}

// ---------------------------------------------------------------------------
// Workspace layout (bytes, 256-aligned):
//   PT   [176,32768] f32 : 0          .. 23068672   (transposed projections)
//   rep  [256,1536]  f32 : 23068672   .. 24641536
//   CS   [256,320]   f32 : 24641536   .. 24969216
//   Vhi  [176,768]  bf16 : 24969216   .. 25239552
//   Vlo  [176,768]  bf16 : 25239552   .. 25509888
//   Whi  [320,1536] bf16 : 25509888   .. 26492928
//   Wlo  [320,1536] bf16 : 26492928   .. 27475968
//   mcnt [256]       i32 : 27475968   .. 27476992
//   dsum [256,168]   f32 : 27476992   .. 27649024  (~27.7 MB total)
// ---------------------------------------------------------------------------
extern "C" void kernel_launch(void* const* d_in, const int* in_sizes, int n_in,
                              void* d_out, int out_size, void* d_ws, size_t ws_size,
                              hipStream_t stream) {
    const float* H    = (const float*)d_in[0];
    const int*   tt   = (const int*)  d_in[1];
    const int*   am   = (const int*)  d_in[2];
    const float* Wc   = (const float*)d_in[3];
    const float* bc   = (const float*)d_in[4];
    const float* Ws   = (const float*)d_in[5];
    const float* bs   = (const float*)d_in[6];
    const float* gate = (const float*)d_in[7];
    const float* ln_g = (const float*)d_in[8];
    const float* ln_b = (const float*)d_in[9];
    const float* Wcls = (const float*)d_in[10];
    const float* bcls = (const float*)d_in[11];
    const float* v8   = (const float*)d_in[12];
    const float* v32  = (const float*)d_in[13];
    const float* v128 = (const float*)d_in[14];
    float* out = (float*)d_out;

    char* ws = (char*)d_ws;
    float*  PT   = (float*) (ws + 0);
    float*  rep  = (float*) (ws + 23068672);
    float*  CS   = (float*) (ws + 24641536);
    __bf16* Vhi  = (__bf16*)(ws + 24969216);
    __bf16* Vlo  = (__bf16*)(ws + 25239552);
    __bf16* Whi  = (__bf16*)(ws + 25509888);
    __bf16* Wlo  = (__bf16*)(ws + 26492928);
    int*    mcnt = (int*)   (ws + 27475968);
    float*  dsum = (float*) (ws + 27476992);

    pack_v_kernel<<<(176 * 768 + 255) / 256, 256, 0, stream>>>(v8, v32, v128, Vhi, Vlo);
    pack_w_kernel<<<(320 * 1536 + 255) / 256, 256, 0, stream>>>(Wc, Ws, Whi, Wlo);
    prep_kernel<<<256, 256, 0, stream>>>(H, tt, am, rep, mcnt);

    // PT^T = H @ Vpad^T : M=32768 (2048 m-tiles), N=176 (11 tiles), K=768 (24 steps)
    wmma_gemm_split_kernel<11, 768, 768, true><<<dim3(1024, 1), 64, 0, stream>>>(
        H, Vhi, Vlo, PT, 2048, 24, (size_t)32768);
    // CS = rep @ W2^T : M=256 (16 m-tiles), N=320 (2 y-blocks x 10 tiles), K=1536
    wmma_gemm_split_kernel<10, 1536, 1536, false><<<dim3(8, 2), 64, 0, stream>>>(
        rep, Whi, Wlo, CS, 16, 48, (size_t)320);

    sort_w1_kernel<<<dim3(168, 256), 128, 0, stream>>>(PT, tt, am, mcnt, dsum);
    finalize_kernel<<<256, 512, 0, stream>>>(CS, bc, bs, gate, ln_g, ln_b,
                                             Wcls, bcls, dsum, mcnt, out);
}
